// GAT_29661044146326
// MI455X (gfx1250) — compile-verified, matrix operand order checked by reading, not run on hardware
//
#include <hip/hip_runtime.h>

typedef _Float16 f16;
typedef __attribute__((ext_vector_type(16))) _Float16 v16h;
typedef __attribute__((ext_vector_type(8)))  float    v8f;

#define N_NODES 50000
#define N_EDGES 800000
#define IN_FEATS 128
#define KDIM 128          // contraction dim for every layer
#define KSTEPS (KDIM / 32)

// ---------------- helpers ----------------

__device__ __forceinline__ void atom_add_f32(float* p, float v) {
  unsafeAtomicAdd(p, v);  // global_atomic_add_f32
}

__device__ __forceinline__ void atom_max_f32(float* p, float v) {
  // IEEE-monotonic integer trick; destination initialized to -inf.
  int vi = __float_as_int(v);
  if (vi >= 0) atomicMax((int*)p, vi);
  else         atomicMin((unsigned int*)p, (unsigned int)vi);
}

// K index inside a 32-wide k-step for WMMA 16-bit fragment element j, lane half `hi`
__device__ __forceinline__ int frag_kidx(int hi, int j) {
  int ko = hi * 8;
  return (j < 8) ? (ko + j) : (16 + ko + (j - 8));
}

// ---------------- operand packing (f32 -> f16, WMMA fragment order) ----------------
// pack[tile][ks][lane][16] ; A: row = lane&15 within m-tile; B: col = lane&15 within n-tile.

__global__ void pack_a_kernel(const float* __restrict__ x, f16* __restrict__ ap,
                              int n_elems, int relu) {
  int i = blockIdx.x * blockDim.x + threadIdx.x;
  int stride = gridDim.x * blockDim.x;
  for (; i < n_elems; i += stride) {
    int j     = i & 15;
    int lane  = (i >> 4) & 31;
    int ks    = (i >> 9) & (KSTEPS - 1);
    int mtile = i >> 11;
    int m  = lane & 15;
    int kidx = ks * 32 + frag_kidx(lane >> 4, j);
    float v = x[(size_t)(mtile * 16 + m) * KDIM + kidx];
    if (relu) v = v > 0.f ? v : 0.f;
    ap[i] = (f16)v;
  }
}

__global__ void pack_b_kernel(const float* __restrict__ w, f16* __restrict__ bp,
                              int n_elems, int Nout) {
  int i = blockIdx.x * blockDim.x + threadIdx.x;
  int stride = gridDim.x * blockDim.x;
  for (; i < n_elems; i += stride) {
    int j     = i & 15;
    int lane  = (i >> 4) & 31;
    int ks    = (i >> 9) & (KSTEPS - 1);
    int ntile = i >> 11;
    int n  = lane & 15;
    int kidx = ks * 32 + frag_kidx(lane >> 4, j);
    bp[i] = (f16)w[(size_t)kidx * Nout + ntile * 16 + n];
  }
}

__global__ void fill_f32_kernel(float* __restrict__ p, int n, float v) {
  int i = blockIdx.x * blockDim.x + threadIdx.x;
  int stride = gridDim.x * blockDim.x;
  for (; i < n; i += stride) p[i] = v;
}

// ---------------- WMMA GEMM: Y[M,Nout] = A @ B + bias, operands pre-packed ----------------
// grid.x = M/16, blockDim.x = (Nout/16)*32; one wave per 16x16 tile; K=128 fully unrolled.

__global__ void wmma_gemm_bias_kernel(const f16* __restrict__ Apack,
                                      const f16* __restrict__ Bpack,
                                      const float* __restrict__ bias,
                                      float* __restrict__ Y,
                                      int Nout) {
  const int lane  = threadIdx.x & 31;
  const int ntile = threadIdx.x >> 5;
  const int mtile = blockIdx.x;
  const int ln = lane & 15;    // C/D column (N); also packed-fragment row id
  const int hi = lane >> 4;

  v8f c;
  const float bv = bias[ntile * 16 + ln];
#pragma unroll
  for (int r = 0; r < 8; ++r) c[r] = bv;

  const v16h* __restrict__ af = (const v16h*)Apack + (size_t)mtile * KSTEPS * 32 + lane;
  const v16h* __restrict__ bf = (const v16h*)Bpack + (size_t)ntile * KSTEPS * 32 + lane;
#pragma unroll
  for (int ks = 0; ks < KSTEPS; ++ks) {
    v16h a = af[ks * 32];
    v16h b = bf[ks * 32];
    c = __builtin_amdgcn_wmma_f32_16x16x32_f16(false, a, false, b,
                                               (short)0, c, false, false);
  }
#pragma unroll
  for (int r = 0; r < 8; ++r) {
    int M = r + 8 * hi;
    Y[(size_t)(mtile * 16 + M) * Nout + ntile * 16 + ln] = c[r];
  }
}

// ---------------- edge kernels (wave per edge, lane = feature f, H heads) ----------------

template <int H>
__global__ void edge_logits_kernel(const float* __restrict__ fs,
                                   const float* __restrict__ fd,
                                   const int* __restrict__ src,
                                   const int* __restrict__ dst,
                                   const float* __restrict__ attn,
                                   float* __restrict__ logits,
                                   float* __restrict__ nmax) {
  int gid = blockIdx.x * blockDim.x + threadIdx.x;
  int e = gid >> 5;
  int lane = gid & 31;
  if (e >= N_EDGES) return;
  const int s = src[e], d = dst[e];
  const int W = H * 32;
  if (e + 16 < N_EDGES) {  // software prefetch of upcoming gathered rows
    __builtin_prefetch(&fs[(size_t)src[e + 16] * W + lane], 0, 1);
    __builtin_prefetch(&fd[(size_t)dst[e + 16] * W + lane], 0, 1);
  }
  float v[H];
#pragma unroll
  for (int h = 0; h < H; ++h) {
    int idx = h * 32 + lane;
    float x = fs[(size_t)s * W + idx] + fd[(size_t)d * W + idx];
    x = x > 0.f ? x : 0.2f * x;               // leaky_relu, slope 0.2
    v[h] = x * attn[idx];
  }
#pragma unroll
  for (int h = 0; h < H; ++h)
#pragma unroll
    for (int off = 16; off; off >>= 1)
      v[h] += __shfl_xor(v[h], off, 32);
  if (lane < H) {
    float lv = v[0];
    if (H == 4) {
      if (lane == 1) lv = v[1];
      else if (lane == 2) lv = v[2];
      else if (lane == 3) lv = v[3];
    }
    logits[(size_t)e * H + lane] = lv;
    atom_max_f32(&nmax[(size_t)d * H + lane], lv);
  }
}

__global__ void edge_exp_kernel(const int* __restrict__ dst,
                                const float* __restrict__ nmax,
                                float* __restrict__ lex,   // in: logits, out: exp
                                float* __restrict__ dsum,
                                int H) {
  int i = blockIdx.x * blockDim.x + threadIdx.x;
  if (i >= N_EDGES * H) return;
  int e = i / H, h = i - e * H;
  int d = dst[e];
  float ex = expf(lex[i] - nmax[(size_t)d * H + h]);
  lex[i] = ex;
  atom_add_f32(&dsum[(size_t)d * H + h], ex);
}

template <int H>
__global__ void edge_aggregate_kernel(const float* __restrict__ fs,
                                      const int* __restrict__ src,
                                      const int* __restrict__ dst,
                                      const float* __restrict__ ex,
                                      const float* __restrict__ dsum,
                                      float* __restrict__ out) {
  int gid = blockIdx.x * blockDim.x + threadIdx.x;
  int e = gid >> 5;
  int lane = gid & 31;
  if (e >= N_EDGES) return;
  const int s = src[e], d = dst[e];
  const int W = H * 32;
  if (e + 16 < N_EDGES)
    __builtin_prefetch(&fs[(size_t)src[e + 16] * W + lane], 0, 1);
#pragma unroll
  for (int h = 0; h < H; ++h) {
    float alpha = ex[(size_t)e * H + h] / dsum[(size_t)d * H + h];
    int idx = h * 32 + lane;
    atom_add_f32(&out[(size_t)d * W + idx], fs[(size_t)s * W + idx] * alpha);
  }
}

// ---------------- host orchestration ----------------

static inline int cdiv(long long a, int b) { return (int)((a + b - 1) / b); }

extern "C" void kernel_launch(void* const* d_in, const int* in_sizes, int n_in,
                              void* d_out, int out_size, void* d_ws, size_t ws_size,
                              hipStream_t stream) {
  const float* features = (const float*)d_in[0];
  const int*   src      = (const int*)d_in[1];
  const int*   dst      = (const int*)d_in[2];
  const float* Ws[3]   = {(const float*)d_in[3],  (const float*)d_in[8],  (const float*)d_in[13]};
  const float* bs[3]   = {(const float*)d_in[4],  (const float*)d_in[9],  (const float*)d_in[14]};
  const float* Wd[3]   = {(const float*)d_in[5],  (const float*)d_in[10], (const float*)d_in[15]};
  const float* bd[3]   = {(const float*)d_in[6],  (const float*)d_in[11], (const float*)d_in[16]};
  const float* attn[3] = {(const float*)d_in[7],  (const float*)d_in[12], (const float*)d_in[17]};
  const int Wsz[3]  = {KDIM * 128, KDIM * 128, KDIM * 32};
  const int Nout[3] = {128, 128, 32};
  const int H[3]    = {4, 4, 1};

  char* ws = (char*)d_ws;
  size_t off = 0;
  auto carve = [&](size_t bytes) -> void* {
    void* p = ws + off;
    off += (bytes + 255) & ~(size_t)255;
    return p;
  };
  f16*   Xh   = (f16*)carve((size_t)N_NODES * KDIM * sizeof(f16));  // packed A fragments
  f16*   Wh[6];
  for (int i = 0; i < 6; ++i) Wh[i] = (f16*)carve((size_t)Wsz[i / 2] * sizeof(f16));
  float* fs   = (float*)carve((size_t)N_NODES * 128 * sizeof(float));
  float* fd   = (float*)carve((size_t)N_NODES * 128 * sizeof(float)); // reused as agg buffer
  float* nmax = (float*)carve((size_t)N_NODES * 4 * sizeof(float));
  float* dsum = (float*)carve((size_t)N_NODES * 4 * sizeof(float));
  float* exb  = (float*)carve((size_t)N_EDGES * 4 * sizeof(float));
  (void)ws_size; (void)n_in; (void)in_sizes;

  const int T = 256;
  // pack operands into WMMA fragment layout (f16)
  pack_a_kernel<<<cdiv((long long)N_NODES * KDIM, T), T, 0, stream>>>(features, Xh,
                                                                      N_NODES * KDIM, 0);
  const float* Wsrc[6] = {Ws[0], Wd[0], Ws[1], Wd[1], Ws[2], Wd[2]};
  const int    Wno[6]  = {128, 128, 128, 128, 32, 32};
  for (int i = 0; i < 6; ++i)
    pack_b_kernel<<<cdiv(Wsz[i / 2], T), T, 0, stream>>>(Wsrc[i], Wh[i], Wsz[i / 2], Wno[i]);

  for (int l = 0; l < 3; ++l) {
    const int h = H[l], no = Nout[l];
    // 1) projections via WMMA: fs = X@Ws+bs, fd = X@Wd+bd
    dim3 gg(N_NODES / 16);
    int gemmT = (no / 16) * 32;
    wmma_gemm_bias_kernel<<<gg, gemmT, 0, stream>>>(Xh, Wh[2 * l],     bs[l], fs, no);
    wmma_gemm_bias_kernel<<<gg, gemmT, 0, stream>>>(Xh, Wh[2 * l + 1], bd[l], fd, no);
    // 2) per-edge logits + segment max
    fill_f32_kernel<<<cdiv((long long)N_NODES * h, T), T, 0, stream>>>(nmax, N_NODES * h, -INFINITY);
    int gridE = cdiv((long long)N_EDGES * 32, T);
    if (h == 4)
      edge_logits_kernel<4><<<gridE, T, 0, stream>>>(fs, fd, src, dst, attn[l], exb, nmax);
    else
      edge_logits_kernel<1><<<gridE, T, 0, stream>>>(fs, fd, src, dst, attn[l], exb, nmax);
    // 3) exp + segment sum
    fill_f32_kernel<<<cdiv((long long)N_NODES * h, T), T, 0, stream>>>(dsum, N_NODES * h, 0.f);
    edge_exp_kernel<<<cdiv((long long)N_EDGES * h, T), T, 0, stream>>>(dst, nmax, exb, dsum, h);
    // 4) scatter-aggregate (fd is dead after step 2; reuse as target, last layer -> d_out)
    float* agg = (l == 2) ? (float*)d_out : fd;
    int aggN = (l == 2) ? out_size : N_NODES * 128;
    fill_f32_kernel<<<cdiv(aggN, T), T, 0, stream>>>(agg, aggN, 0.f);
    if (h == 4)
      edge_aggregate_kernel<4><<<gridE, T, 0, stream>>>(fs, src, dst, exb, dsum, agg);
    else
      edge_aggregate_kernel<1><<<gridE, T, 0, stream>>>(fs, src, dst, exb, dsum, agg);
    // 5) inter-layer ReLU + repack to A fragments for next layer's GEMMs
    if (l < 2)
      pack_a_kernel<<<cdiv((long long)N_NODES * KDIM, T), T, 0, stream>>>(agg, Xh,
                                                                          N_NODES * KDIM, 1);
  }
}
